// AttentionQFunction_40003325395313
// MI455X (gfx1250) — matrix operand order repaired and checked by reading, not compiled, not run on hardware
//
#include <hip/hip_runtime.h>
#include <hip/hip_fp16.h>

typedef __attribute__((ext_vector_type(16))) _Float16 v16h;
typedef __attribute__((ext_vector_type(8)))  _Float16 v8h;
typedef __attribute__((ext_vector_type(4)))  _Float16 v4h;
typedef __attribute__((ext_vector_type(8)))  float    v8f;
typedef __attribute__((ext_vector_type(4)))  float    v4f;
typedef __attribute__((ext_vector_type(4)))  unsigned int u32x4;
typedef __attribute__((ext_vector_type(8)))  int      i32x8;
typedef __attribute__((ext_vector_type(4)))  int      i32x4;

#define B_   256
#define N_   512
#define H_   256
#define ROWS (B_ * N_)   // 131072

#if defined(__has_builtin)
#if __has_builtin(__builtin_amdgcn_tensor_load_to_lds)
#define USE_TDM 1
#else
#define USE_TDM 0
#endif
#else
#define USE_TDM 0
#endif

// ---------------------------------------------------------------------------
// Weight transpose + f16 convert + K-padding:  W[K][N] (f32) -> Wt[N][Kpad] (f16)
// ---------------------------------------------------------------------------
__global__ void transpose_pad(const float* __restrict__ src, _Float16* __restrict__ dst,
                              int K, int N, int Kpad) {
    int total = N * Kpad;
    for (int i = blockIdx.x * blockDim.x + threadIdx.x; i < total;
         i += gridDim.x * blockDim.x) {
        int n = i / Kpad, k = i % Kpad;
        float v = (k < K) ? src[(size_t)k * N + n] : 0.0f;
        dst[i] = (_Float16)v;
    }
}

// ---------------------------------------------------------------------------
// Build X[B*N][32] f16: cols 0..11 = obs (broadcast over n), 12..15 = obstacle
// channels 0..3, 16..31 = zero padding.
// ---------------------------------------------------------------------------
__global__ void build_x(const float* __restrict__ obs, const float* __restrict__ obstacles,
                        _Float16* __restrict__ X) {
    const int total = ROWS * 32;
    for (int i = blockIdx.x * blockDim.x + threadIdx.x; i < total;
         i += gridDim.x * blockDim.x) {
        int row = i >> 5, c = i & 31;
        int b = row >> 9, n = row & 511;
        float v = 0.0f;
        if (c < 12)      v = obs[b * 12 + c];
        else if (c < 16) v = obstacles[(size_t)b * 2560 + (c - 12) * 512 + n];
        X[i] = (_Float16)v;
    }
}

__global__ void zero_f32(float* __restrict__ p, int n) {
    for (int i = blockIdx.x * blockDim.x + threadIdx.x; i < n;
         i += gridDim.x * blockDim.x) p[i] = 0.0f;
}

// ---------------------------------------------------------------------------
// Generic WMMA GEMM:  C[rows][nout] (f16) = act( A[rows][kpad] @ Wt[nout][kpad]^T + bias )
// Block = 128 threads (4 waves). Tile 64 rows x 64 cols; wave w owns rows
// [blk*64 + w*16, +16) and all 4 col tiles.  storeT=1 writes C[nout][rows].
// The 64 x kpad weight slab shared by all 4 waves is staged into LDS by the
// Tensor Data Mover (one descriptor per block, TENSORcnt-tracked).
// ---------------------------------------------------------------------------
__global__ __launch_bounds__(128)
void gemm_f16(const _Float16* __restrict__ A, const _Float16* __restrict__ Wt,
              const float* __restrict__ bias, _Float16* __restrict__ C,
              int rows, int nout, int kpad, int relu, int storeT) {
    __shared__ __align__(16) _Float16 tile[4][4][16][16];   // [wave][t][col][row]
#if USE_TDM
    __shared__ __align__(16) _Float16 wtile[64 * 288];      // weight slab (<=36KB)
#endif
    const int wave = threadIdx.x >> 5;
    const int lane = threadIdx.x & 31;
    const int l16  = lane & 15;
    const int half = lane >> 4;
    const int row0 = blockIdx.x * 64 + wave * 16;
    const int col0 = blockIdx.y * 64;

#if USE_TDM
    if (threadIdx.x < 32) {
        // Flat 1-D contiguous tile: 64*kpad f16 elements (<= 18432, fits tile_dim0).
        unsigned long long gaddr = (unsigned long long)(const void*)(Wt + (size_t)col0 * kpad);
        unsigned int laddr = (unsigned int)(unsigned long long)(const void*)&wtile[0];
        unsigned int nelem = (unsigned int)(64 * kpad);
        u32x4 g0;
        g0[0] = 1u;                                          // count=1, is_restore=0
        g0[1] = laddr;                                       // lds_addr
        g0[2] = (unsigned int)gaddr;                         // global_addr[31:0]
        g0[3] = (unsigned int)(gaddr >> 32) | 0x80000000u;   // global_addr[56:32] | type=2
        i32x8 g1;
        g1[0] = 0x00010000;                                  // data_size=1 (2 bytes)
        g1[1] = (int)((nelem & 0xFFFFu) << 16);              // tensor_dim0[15:0] @ bits 63:48
        g1[2] = (int)((nelem >> 16) | (1u << 16));           // tensor_dim0[31:16] | tensor_dim1=1
        g1[3] = (int)((nelem & 0xFFFFu) << 16);              // tensor_dim1[31:16]=0 | tile_dim0
        g1[4] = 1;                                           // tile_dim1=1 | tile_dim2=0
        g1[5] = (int)nelem;                                  // tensor_dim0_stride[31:0]
        g1[6] = 0;                                           // stride hi | dim1_stride lo
        g1[7] = 0;
        i32x4 g2 = {0, 0, 0, 0};
        i32x4 g3 = {0, 0, 0, 0};
#if defined(__clang_major__) && (__clang_major__ >= 23)
        i32x8 g4 = {0, 0, 0, 0, 0, 0, 0, 0};
        __builtin_amdgcn_tensor_load_to_lds(g0, g1, g2, g3, g4, 0);
#else
        __builtin_amdgcn_tensor_load_to_lds(g0, g1, g2, g3, 0);
#endif
#if __has_builtin(__builtin_amdgcn_s_wait_tensorcnt)
        __builtin_amdgcn_s_wait_tensorcnt(0);
#else
        asm volatile("s_wait_tensorcnt 0x0" ::: "memory");
#endif
    }
    __syncthreads();
#endif

    v8f acc[4] = {};
    const _Float16* arow = A + (size_t)(row0 + l16) * kpad + half * 16;

    for (int kc = 0; kc < kpad; kc += 32) {
        if (kc + 32 < kpad) __builtin_prefetch(arow + kc + 32);
        v8h alo = *(const v8h*)(arow + kc);
        v8h ahi = *(const v8h*)(arow + kc + 8);
        v16h af = __builtin_shufflevector(alo, ahi,
                     0,1,2,3,4,5,6,7,8,9,10,11,12,13,14,15);
#pragma unroll
        for (int t = 0; t < 4; ++t) {
#if USE_TDM
            const _Float16* wp = wtile + (size_t)(t * 16 + l16) * kpad + kc + half * 16;
#else
            const _Float16* wp = Wt + (size_t)(col0 + t * 16 + l16) * kpad + kc + half * 16;
#endif
            v8h blo = *(const v8h*)wp;
            v8h bhi = *(const v8h*)(wp + 8);
            v16h bf = __builtin_shufflevector(blo, bhi,
                         0,1,2,3,4,5,6,7,8,9,10,11,12,13,14,15);
            acc[t] = __builtin_amdgcn_wmma_f32_16x16x32_f16(
                         false, af, false, bf, (short)0, acc[t], false, false);
        }
    }

    if (storeT) {
        // D layout: lane owns col (l16) and 8 consecutive rows (half*8 + j).
#pragma unroll
        for (int t = 0; t < 4; ++t) {
            int n = col0 + t * 16 + l16;
            float bv = bias[n];
            v8h hv;
#pragma unroll
            for (int j = 0; j < 8; ++j) {
                float x = acc[t][j] + bv;
                if (relu) x = fmaxf(x, 0.0f);
                hv[j] = (_Float16)x;
            }
            *(v8h*)(C + (size_t)n * rows + row0 + half * 8) = hv;
        }
    } else {
        // LDS transpose so global stores are coalesced b128 along the row dim.
#pragma unroll
        for (int t = 0; t < 4; ++t) {
            int n = col0 + t * 16 + l16;
            float bv = bias[n];
            v8h hv;
#pragma unroll
            for (int j = 0; j < 8; ++j) {
                float x = acc[t][j] + bv;
                if (relu) x = fmaxf(x, 0.0f);
                hv[j] = (_Float16)x;
            }
            *(v8h*)&tile[wave][t][l16][half * 8] = hv;
        }
        __syncthreads();
#pragma unroll
        for (int t = 0; t < 4; ++t) {
            v8h ov;
#pragma unroll
            for (int j = 0; j < 8; ++j) ov[j] = tile[wave][t][half * 8 + j][l16];
            *(v8h*)(C + (size_t)(row0 + l16) * nout + col0 + t * 16 + half * 8) = ov;
        }
    }
}

// ---------------------------------------------------------------------------
// Fused attention per (batch, 64-row block): S = QK^T*scale masked -> softmax
// -> O = P V -> row-mask -> atomic max-reduce over rows into outmax[b][d].
// Block = 128 threads (4 waves); wave owns 16 rows.
// Dynamic LDS: S f32 [4][16][512] (128KB) + P f16 [4][16][512] (64KB) + rowsum.
// ---------------------------------------------------------------------------
__global__ __launch_bounds__(128)
void attn_kernel(const _Float16* __restrict__ Q, const _Float16* __restrict__ Kmat,
                 const _Float16* __restrict__ Vt, const float* __restrict__ obstacles,
                 float* __restrict__ outmax) {
    extern __shared__ char lds[];
    const int b    = blockIdx.x;
    const int rb   = blockIdx.y;
    const int wave = threadIdx.x >> 5;
    const int lane = threadIdx.x & 31;
    const int l16  = lane & 15;
    const int half = lane >> 4;
    const int r0   = rb * 64 + wave * 16;

    float*    S      = (float*)lds + (size_t)wave * 16 * 512;
    _Float16* P      = (_Float16*)(lds + 131072) + (size_t)wave * 16 * 512;
    float*    rowsum = (float*)(lds + 131072 + 65536) + wave * 16;

    const float* mask = obstacles + (size_t)b * 2560 + 4 * 512;

    // Preload this wave's 16 Q rows as 8 A-fragments (K = 256).
    const _Float16* qrow = Q + (size_t)(b * 512 + r0 + l16) * 256 + half * 16;
    v16h qf[8];
#pragma unroll
    for (int kc = 0; kc < 8; ++kc) {
        v8h lo = *(const v8h*)(qrow + kc * 32);
        v8h hi = *(const v8h*)(qrow + kc * 32 + 8);
        qf[kc] = __builtin_shufflevector(lo, hi,
                    0,1,2,3,4,5,6,7,8,9,10,11,12,13,14,15);
    }

    // ---- Phase 1: S[16][512] = Q . K^T * scale, column-masked ----
    for (int m0 = 0; m0 < 512; m0 += 16) {
        v8f acc = {};
        const _Float16* krow = Kmat + (size_t)(b * 512 + m0 + l16) * 256 + half * 16;
#pragma unroll
        for (int kc = 0; kc < 8; ++kc) {
            v8h blo = *(const v8h*)(krow + kc * 32);
            v8h bhi = *(const v8h*)(krow + kc * 32 + 8);
            v16h bf = __builtin_shufflevector(blo, bhi,
                        0,1,2,3,4,5,6,7,8,9,10,11,12,13,14,15);
            acc = __builtin_amdgcn_wmma_f32_16x16x32_f16(
                      false, qf[kc], false, bf, (short)0, acc, false, false);
        }
        float mk = mask[m0 + l16];
        float* Sp = S + (half * 8) * 512 + (m0 + l16);
#pragma unroll
        for (int j = 0; j < 8; ++j) {
            float sv = (mk == 0.0f) ? -1.0e9f : acc[j] * 0.0625f; // 1/sqrt(256)
            Sp[j * 512] = sv;
        }
    }
    __syncthreads();

    // ---- Softmax over 512 cols; lane pair (l16, l16+16) splits the row ----
    {
        const float* Sr = S + l16 * 512 + half * 256;
        float mx = -3.0e38f;
        for (int c = 0; c < 256; c += 4) {
            v4f v = *(const v4f*)(Sr + c);
            mx = fmaxf(mx, fmaxf(fmaxf(v[0], v[1]), fmaxf(v[2], v[3])));
        }
        mx = fmaxf(mx, __shfl_xor(mx, 16));
        _Float16* Pr = P + l16 * 512 + half * 256;
        float sum = 0.0f;
        for (int c = 0; c < 256; c += 4) {
            v4f v = *(const v4f*)(Sr + c);
            float e0 = __expf(v[0] - mx), e1 = __expf(v[1] - mx);
            float e2 = __expf(v[2] - mx), e3 = __expf(v[3] - mx);
            sum += (e0 + e1) + (e2 + e3);
            v4h h; h[0] = (_Float16)e0; h[1] = (_Float16)e1;
                   h[2] = (_Float16)e2; h[3] = (_Float16)e3;
            *(v4h*)(Pr + c) = h;
        }
        sum += __shfl_xor(sum, 16);
        if (half == 0) rowsum[l16] = sum;
    }
    __syncthreads();

    // ---- Phase 2: O[16][256] = P . V  (Vt is [256][ROWS], per-batch slice) ----
    v8f o[16] = {};
    const _Float16* pbase = P + l16 * 512;
    for (int m0 = 0; m0 < 512; m0 += 32) {
        v8h alo = *(const v8h*)(pbase + m0 + half * 16);
        v8h ahi = *(const v8h*)(pbase + m0 + 8 + half * 16);
        v16h af = __builtin_shufflevector(alo, ahi,
                    0,1,2,3,4,5,6,7,8,9,10,11,12,13,14,15);
#pragma unroll
        for (int t = 0; t < 16; ++t) {
            const _Float16* vp = Vt + (size_t)(t * 16 + l16) * ROWS
                                    + (size_t)b * 512 + m0 + half * 16;
            v8h blo = *(const v8h*)vp;
            v8h bhi = *(const v8h*)(vp + 8);
            v16h bf = __builtin_shufflevector(blo, bhi,
                        0,1,2,3,4,5,6,7,8,9,10,11,12,13,14,15);
            o[t] = __builtin_amdgcn_wmma_f32_16x16x32_f16(
                       false, af, false, bf, (short)0, o[t], false, false);
        }
    }

    // ---- Epilogue: normalize, row-mask, max over rows, atomic into outmax ----
    const float* maskn = obstacles + (size_t)b * 2560 + 4 * 512 + r0 + half * 8;
    float rs[8], mr[8];
#pragma unroll
    for (int j = 0; j < 8; ++j) { rs[j] = rowsum[half * 8 + j]; mr[j] = maskn[j]; }
#pragma unroll
    for (int t = 0; t < 16; ++t) {
        int d = t * 16 + l16;
        float mx = 0.0f;
#pragma unroll
        for (int j = 0; j < 8; ++j)
            mx = fmaxf(mx, o[t][j] * mr[j] / rs[j]);
        mx = fmaxf(mx, __shfl_xor(mx, 16));
        if (half == 0)
            atomicMax((int*)(outmax + b * 256 + d), __float_as_int(mx));
    }
}

// ---------------------------------------------------------------------------
// comb[256][288] f16 = [obs(12) | outmax(256) | act(2) | pad(18)]
// ---------------------------------------------------------------------------
__global__ void build_comb(const float* __restrict__ obs, const float* __restrict__ outmax,
                           const float* __restrict__ act, _Float16* __restrict__ comb) {
    int i = blockIdx.x * blockDim.x + threadIdx.x;
    if (i >= 256 * 288) return;
    int b = i / 288, c = i % 288;
    float v = 0.0f;
    if (c < 12)       v = obs[b * 12 + c];
    else if (c < 268) v = outmax[b * 256 + (c - 12)];
    else if (c < 270) v = act[b * 2 + (c - 268)];
    comb[i] = (_Float16)v;
}

// Final layer: qv[b] = dot(H[b][0..256], W) + bias
__global__ __launch_bounds__(256)
void head_final(const _Float16* __restrict__ H, const float* __restrict__ W,
                const float* __restrict__ bf, float* __restrict__ out) {
    __shared__ float red[256];
    int b = blockIdx.x, t = threadIdx.x;
    red[t] = (float)H[(size_t)b * 256 + t] * W[t];
    __syncthreads();
    for (int s = 128; s > 0; s >>= 1) {
        if (t < s) red[t] += red[t + s];
        __syncthreads();
    }
    if (t == 0) out[b] = red[0] + bf[0];
}

// ---------------------------------------------------------------------------
extern "C" void kernel_launch(void* const* d_in, const int* in_sizes, int n_in,
                              void* d_out, int out_size, void* d_ws, size_t ws_size,
                              hipStream_t stream) {
    (void)in_sizes; (void)n_in; (void)out_size; (void)ws_size;
    const float* obs       = (const float*)d_in[0];
    const float* obstacles = (const float*)d_in[1];
    const float* act       = (const float*)d_in[2];
    const float* qW0 = (const float*)d_in[3],  *qb0 = (const float*)d_in[4];
    const float* qW1 = (const float*)d_in[5],  *qb1 = (const float*)d_in[6];
    const float* qW2 = (const float*)d_in[7],  *qb2 = (const float*)d_in[8];
    const float* kW0 = (const float*)d_in[9],  *kb0 = (const float*)d_in[10];
    const float* kW1 = (const float*)d_in[11], *kb1 = (const float*)d_in[12];
    const float* kW2 = (const float*)d_in[13], *kb2 = (const float*)d_in[14];
    const float* vW0 = (const float*)d_in[15], *vb0 = (const float*)d_in[16];
    const float* vW1 = (const float*)d_in[17], *vb1 = (const float*)d_in[18];
    const float* vW2 = (const float*)d_in[19], *vb2 = (const float*)d_in[20];
    const float* hW0 = (const float*)d_in[21], *hb0 = (const float*)d_in[22];
    const float* hW1 = (const float*)d_in[23], *hb1 = (const float*)d_in[24];
    const float* hW2 = (const float*)d_in[25], *hb2 = (const float*)d_in[26];

    char* ws = (char*)d_ws;
    size_t off = 0;
    auto alloc = [&](size_t bytes) -> char* {
        off = (off + 255) & ~(size_t)255;
        char* p = ws + off; off += bytes; return p;
    };

    _Float16* wq0t = (_Float16*)alloc(256 * 32 * 2);
    _Float16* wq1t = (_Float16*)alloc(65536 * 2);
    _Float16* wq2t = (_Float16*)alloc(65536 * 2);
    _Float16* wk0t = (_Float16*)alloc(256 * 32 * 2);
    _Float16* wk1t = (_Float16*)alloc(65536 * 2);
    _Float16* wk2t = (_Float16*)alloc(65536 * 2);
    _Float16* wv0t = (_Float16*)alloc(256 * 32 * 2);
    _Float16* wv1t = (_Float16*)alloc(65536 * 2);
    _Float16* wv2t = (_Float16*)alloc(65536 * 2);
    _Float16* wh0t = (_Float16*)alloc(256 * 288 * 2);
    _Float16* wh1t = (_Float16*)alloc(65536 * 2);
    _Float16* X    = (_Float16*)alloc((size_t)ROWS * 32 * 2);
    _Float16* Abuf = (_Float16*)alloc((size_t)ROWS * 256 * 2);
    _Float16* Bbuf = (_Float16*)alloc((size_t)ROWS * 256 * 2);
    _Float16* Qb   = (_Float16*)alloc((size_t)ROWS * 256 * 2);
    _Float16* Kb   = (_Float16*)alloc((size_t)ROWS * 256 * 2);
    _Float16* Vt   = (_Float16*)alloc((size_t)ROWS * 256 * 2);
    float*    omax = (float*)alloc(65536 * 4);
    _Float16* comb = (_Float16*)alloc(256 * 288 * 2);
    _Float16* Hh1  = (_Float16*)alloc(65536 * 2);
    _Float16* Hh2  = (_Float16*)alloc(65536 * 2);

    // Weight prep
    transpose_pad<<<128, 256, 0, stream>>>(qW0, wq0t, 16, 256, 32);
    transpose_pad<<<256, 256, 0, stream>>>(qW1, wq1t, 256, 256, 256);
    transpose_pad<<<256, 256, 0, stream>>>(qW2, wq2t, 256, 256, 256);
    transpose_pad<<<128, 256, 0, stream>>>(kW0, wk0t, 16, 256, 32);
    transpose_pad<<<256, 256, 0, stream>>>(kW1, wk1t, 256, 256, 256);
    transpose_pad<<<256, 256, 0, stream>>>(kW2, wk2t, 256, 256, 256);
    transpose_pad<<<128, 256, 0, stream>>>(vW0, wv0t, 16, 256, 32);
    transpose_pad<<<256, 256, 0, stream>>>(vW1, wv1t, 256, 256, 256);
    transpose_pad<<<256, 256, 0, stream>>>(vW2, wv2t, 256, 256, 256);
    transpose_pad<<<288, 256, 0, stream>>>(hW0, wh0t, 270, 256, 288);
    transpose_pad<<<256, 256, 0, stream>>>(hW1, wh1t, 256, 256, 256);

    build_x<<<2048, 256, 0, stream>>>(obs, obstacles, X);
    zero_f32<<<64, 256, 0, stream>>>(omax, 65536);

    dim3 gBig(ROWS / 64, 4);
    // Q
    gemm_f16<<<gBig, 128, 0, stream>>>(X,    wq0t, qb0, Abuf, ROWS, 256, 32,  1, 0);
    gemm_f16<<<gBig, 128, 0, stream>>>(Abuf, wq1t, qb1, Bbuf, ROWS, 256, 256, 1, 0);
    gemm_f16<<<gBig, 128, 0, stream>>>(Bbuf, wq2t, qb2, Qb,   ROWS, 256, 256, 0, 0);
    // K
    gemm_f16<<<gBig, 128, 0, stream>>>(X,    wk0t, kb0, Abuf, ROWS, 256, 32,  1, 0);
    gemm_f16<<<gBig, 128, 0, stream>>>(Abuf, wk1t, kb1, Bbuf, ROWS, 256, 256, 1, 0);
    gemm_f16<<<gBig, 128, 0, stream>>>(Bbuf, wk2t, kb2, Kb,   ROWS, 256, 256, 0, 0);
    // V (out_act = relu, stored transposed [256][ROWS])
    gemm_f16<<<gBig, 128, 0, stream>>>(X,    wv0t, vb0, Abuf, ROWS, 256, 32,  1, 0);
    gemm_f16<<<gBig, 128, 0, stream>>>(Abuf, wv1t, vb1, Bbuf, ROWS, 256, 256, 1, 0);
    gemm_f16<<<gBig, 128, 0, stream>>>(Bbuf, wv2t, vb2, Vt,   ROWS, 256, 256, 1, 1);

    // Attention: grid (batch, row-block), 192KB+256B dynamic LDS
    attn_kernel<<<dim3(B_, 8), 128, 131072 + 65536 + 256, stream>>>(
        Qb, Kb, Vt, obstacles, omax);

    build_comb<<<288, 256, 0, stream>>>(obs, omax, act, comb);

    dim3 gHead(4, 4);
    gemm_f16<<<gHead, 128, 0, stream>>>(comb, wh0t, hb0, Hh1, 256, 256, 288, 1, 0);
    gemm_f16<<<gHead, 128, 0, stream>>>(Hh1,  wh1t, hb1, Hh2, 256, 256, 256, 1, 0);
    head_final<<<256, 256, 0, stream>>>(Hh2, hW2, hb2, (float*)d_out);
}
